// dualGCN_77086073029158
// MI455X (gfx1250) — compile-verified, matrix operand order checked by reading, not run on hardware
//
#include <hip/hip_runtime.h>
#include <hip/hip_fp16.h>

typedef __attribute__((ext_vector_type(16))) _Float16 v16h;
typedef __attribute__((ext_vector_type(8)))  float    v8f;

#define NN   50000
#define F_IN 128
#define HID  128
#define LOUT 64

// ---------------- degree / norm ----------------
__global__ void k_init_deg(float* deg, int n) {
    int i = blockIdx.x * blockDim.x + threadIdx.x;
    if (i < n) deg[i] = 1.0f;                      // self-loop contribution
}
__global__ void k_acc_deg(const int* __restrict__ dst, float* deg, int E) {
    int e = blockIdx.x * blockDim.x + threadIdx.x;
    if (e < E) atomicAdd(&deg[dst[e]], 1.0f);
}
__global__ void k_rsqrt(float* deg, int n) {
    int i = blockIdx.x * blockDim.x + threadIdx.x;
    if (i < n) deg[i] = rsqrtf(deg[i]);            // deg >= 1 always
}

// ---------------- conversions ----------------
__global__ void k_f32_to_f16(const float* __restrict__ x, _Float16* __restrict__ y, int n) {
    int i = blockIdx.x * blockDim.x + threadIdx.x;
    if (i < n) y[i] = (_Float16)x[i];
}
__global__ void k_relu_f16(const float* __restrict__ a, _Float16* __restrict__ y, int n) {
    int i = blockIdx.x * blockDim.x + threadIdx.x;
    if (i < n) y[i] = (_Float16)fmaxf(a[i], 0.0f);
}

// Pack W[fi x fo] (row-major f32) into WMMA B-operand lane layout, f16.
// Bp index = ((kt*ntiles + nt)*32 + lane)*16 + e
// element e of lane: n = nt*16 + (lane&15); k = kt*32 + (lane>>4)*8 + (e&7) + (e&8 ? 16 : 0)
__global__ void k_pack_w(const float* __restrict__ W, _Float16* __restrict__ Bp,
                         int fo, int total) {
    int idx = blockIdx.x * blockDim.x + threadIdx.x;
    if (idx >= total) return;
    int e    = idx & 15;
    int lane = (idx >> 4) & 31;
    int tile = idx >> 9;
    int ntiles = fo >> 4;
    int nt = tile % ntiles;
    int kt = tile / ntiles;
    int ncol = (nt << 4) + (lane & 15);
    int k = (kt << 5) + ((lane >> 4) << 3) + (e & 7) + ((e & 8) ? 16 : 0);
    Bp[idx] = (_Float16)W[k * fo + ncol];
}

// ---------------- WMMA GEMM: C[N x fo] = A[N x K](f16) @ Bp(packed f16), f32 accum ----
__global__ __launch_bounds__(256) void k_gemm_wmma(
    const _Float16* __restrict__ A, const _Float16* __restrict__ Bp,
    float* __restrict__ C, int mtiles, int K, int fo) {
    int wave = (int)((blockIdx.x * blockDim.x + threadIdx.x) >> 5);
    int lane = threadIdx.x & 31;
    int ntiles = fo >> 4;
    int mt = wave / ntiles;
    int nt = wave - mt * ntiles;
    if (mt >= mtiles) return;                       // wave-uniform: EXEC stays all-ones
    int half = lane >> 4;
    const _Float16* arow = A + (size_t)(((mt << 4) + (lane & 15)) * K + (half << 3));
    const _Float16* bptr = Bp + (size_t)nt * 512 + (size_t)lane * 16;
    v8f acc = {};
    int ktiles = K >> 5;
    for (int kt = 0; kt < ktiles; ++kt) {
        union { uint4 u[2]; v16h v; } ua, ub;
        ua.u[0] = *(const uint4*)(arow + (kt << 5));
        ua.u[1] = *(const uint4*)(arow + (kt << 5) + 16);
        ub.u[0] = *(const uint4*)(bptr + (size_t)kt * ntiles * 512);
        ub.u[1] = *(const uint4*)(bptr + (size_t)kt * ntiles * 512 + 8);
        acc = __builtin_amdgcn_wmma_f32_16x16x32_f16(
            false, ua.v, false, ub.v, (short)0, acc, false, false);
    }
    int col = (nt << 4) + (lane & 15);
    int rowbase = (mt << 4) + (half << 3);
#pragma unroll
    for (int r = 0; r < 8; ++r)
        C[(size_t)(rowbase + r) * fo + col] = acc[r];
}

// ---------------- aggregation ----------------
// agg[i,j] = bias[j] + dinv[i]^2 * xw[i,j]   (self-loop term + bias)
__global__ void k_init_agg(float* __restrict__ agg, const float* __restrict__ xw,
                           const float* __restrict__ bias, const float* __restrict__ dinv,
                           int total, int foShift) {
    int idx = blockIdx.x * blockDim.x + threadIdx.x;
    if (idx >= total) return;
    int fo = 1 << foShift;
    int i = idx >> foShift;
    int j = idx & (fo - 1);
    float d = dinv[i];
    agg[idx] = bias[j] + d * d * xw[idx];
}

// agg[dst,j] += dinv[src]*dinv[dst] * xw[src,j]   — coalesced gather + f32 atomics
__global__ void k_scatter(float* __restrict__ agg, const float* __restrict__ xw,
                          const int* __restrict__ src, const int* __restrict__ dst,
                          const float* __restrict__ dinv, long total, int foShift) {
    long t = (long)blockIdx.x * blockDim.x + threadIdx.x;
    if (t >= total) return;
    int fo = 1 << foShift;
    int e = (int)(t >> foShift);
    int j = (int)(t & (fo - 1));
    int s = src[e], d = dst[e];
    float nrm = dinv[s] * dinv[d];
    atomicAdd(&agg[((size_t)d << foShift) + j], nrm * xw[((size_t)s << foShift) + j]);
}

// out = (or +=) 0.5*sigmoid(agg)
__global__ void k_sigmoid_out(const float* __restrict__ a, float* __restrict__ out,
                              int n, int first) {
    int i = blockIdx.x * blockDim.x + threadIdx.x;
    if (i >= n) return;
    float v = 0.5f / (1.0f + expf(-a[i]));
    if (first) out[i] = v;
    else       out[i] += v;
}

// ---------------- driver ----------------
extern "C" void kernel_launch(void* const* d_in, const int* in_sizes, int n_in,
                              void* d_out, int out_size, void* d_ws, size_t ws_size,
                              hipStream_t stream) {
    const int N = NN;
    const int E = in_sizes[1] / 2;

    char* ws = (char*)d_ws;
    float*    dinv = (float*)(ws + 0);            //  N*4           = 200000  (pad 200704)
    _Float16* h16  = (_Float16*)(ws + 200704);    //  N*128*2       = 12.8 MB
    float*    xw   = (float*)(ws + 13000704);     //  N*128*4       = 25.6 MB
    float*    agg  = (float*)(ws + 38600704);     //  N*128*4       = 25.6 MB
    _Float16* wpk  = (_Float16*)(ws + 64200704);  //  57344 halves  = 112 KB
    float* out = (float*)d_out;

    const int fi_l[4]   = {F_IN, HID, HID, HID};
    const int fo_l[4]   = {HID, HID, HID, LOUT};
    const int foSh_l[4] = {7, 7, 7, 6};
    const int woff[4]   = {0, 16384, 32768, 49152};   // halves

    for (int b = 0; b < 2; ++b) {
        const float* x  = (const float*)d_in[2 * b];
        const int*   ei = (const int*)d_in[1 + 2 * b];
        const int* srcI = ei;
        const int* dstI = ei + E;

        // symmetric-normalization factors
        k_init_deg<<<(N + 255) / 256, 256, 0, stream>>>(dinv, N);
        k_acc_deg<<<(E + 255) / 256, 256, 0, stream>>>(dstI, dinv, E);
        k_rsqrt<<<(N + 255) / 256, 256, 0, stream>>>(dinv, N);

        // first-layer input in f16
        k_f32_to_f16<<<(N * F_IN + 255) / 256, 256, 0, stream>>>(x, h16, N * F_IN);

        // pack all 4 weight matrices into WMMA B layout
        for (int l = 0; l < 4; ++l) {
            const float* W = (const float*)d_in[4 + 8 * b + 2 * l];
            int tot = fi_l[l] * fo_l[l];
            k_pack_w<<<(tot + 255) / 256, 256, 0, stream>>>(W, wpk + woff[l], fo_l[l], tot);
        }

        for (int l = 0; l < 4; ++l) {
            int K = fi_l[l], fo = fo_l[l], foSh = foSh_l[l];
            const float* bias = (const float*)d_in[4 + 8 * b + 2 * l + 1];

            int waves = (N / 16) * (fo / 16);
            k_gemm_wmma<<<(waves * 32 + 255) / 256, 256, 0, stream>>>(
                h16, wpk + woff[l], xw, N / 16, K, fo);

            int tot = N * fo;
            k_init_agg<<<(tot + 255) / 256, 256, 0, stream>>>(agg, xw, bias, dinv, tot, foSh);

            long etot = (long)E * fo;
            k_scatter<<<(int)((etot + 255) / 256), 256, 0, stream>>>(
                agg, xw, srcI, dstI, dinv, etot, foSh);

            if (l < 3)
                k_relu_f16<<<(tot + 255) / 256, 256, 0, stream>>>(agg, h16, tot);
            else
                k_sigmoid_out<<<(tot + 255) / 256, 256, 0, stream>>>(agg, out, tot, b == 0 ? 1 : 0);
        }
    }
}
